// Attention_75239237091669
// MI455X (gfx1250) — matrix operand order repaired and checked by reading, not compiled, not run on hardware
//
#include <hip/hip_runtime.h>
#include <hip/hip_bf16.h>
#include <stdint.h>

// ---------------------------------------------------------------------------
// CDNA5 / gfx1250 wave32 WMMA attention pipeline (f16 matrix cores, f32 accum)
// + Tensor Data Mover double-buffered K staging in the attention kernel.
// ---------------------------------------------------------------------------

typedef _Float16 v16h __attribute__((ext_vector_type(16)));
typedef _Float16 v8h  __attribute__((ext_vector_type(8)));
typedef float    v8f  __attribute__((ext_vector_type(8)));
typedef unsigned int u32x4 __attribute__((ext_vector_type(4)));
typedef int          i32x4 __attribute__((ext_vector_type(4)));
typedef int          i32x8 __attribute__((ext_vector_type(8)));

union U16H { v16h v; v8h h[2]; _Float16 e[16]; };
union U8H  { v8h v; _Float16 e[8]; };
union U8F  { v8f v; float e[8]; };

#define BQ   2
#define TQ   1024
#define DQ   2560
#define NQ   32
#define KHQ  8
#define HQ   128
#define K_MASK_F (-2.3819763e38f)

#if defined(__HIP_DEVICE_COMPILE__) && \
    __has_builtin(__builtin_amdgcn_tensor_load_to_lds) && \
    __has_builtin(__builtin_amdgcn_s_wait_tensorcnt)
#define USE_TDM 1
#else
#define USE_TDM 0
#endif

// ---------------------------------------------------------------------------
// GEMM: C[M,N] = A[M,K]*B[K,N], fp32 in/out, f16 WMMA compute.
// 256 threads (8 waves). Block tile 64x64, wave tile 16x32 (2 WMMA / K-step).
// A staged row-major (vector ds_store_b128), B staged transposed so both
// fragments are contiguous 128-bit LDS loads.
// ---------------------------------------------------------------------------
__global__ __launch_bounds__(256) void gemm_wmma_f16(
    const float* __restrict__ A, const float* __restrict__ B,
    float* __restrict__ C, int M, int N, int K)
{
  __shared__ __align__(32) _Float16 As[64][48];    // [m][k], rows 96B
  __shared__ __align__(32) _Float16 Bst[64][48];   // [n][k], transposed

  const int tid  = threadIdx.x;
  const int lane = tid & 31;
  const int wv   = tid >> 5;            // 0..7
  const int col  = lane & 15;
  const int hf   = lane >> 4;           // lane half
  const int wr   = (wv >> 1) * 16;      // wave row: 0,16,32,48
  const int wc   = (wv & 1) * 32;       // wave col: 0,32
  const int bm   = blockIdx.y * 64;
  const int bn   = blockIdx.x * 64;

  const int arow = tid >> 2;            // 0..63
  const int acol = (tid & 3) * 8;       // 0,8,16,24
  const int brow = tid >> 3;            // 0..31 (k)
  const int bcol = (tid & 7) * 8;       // 0..56 (n)

  U8F acc0, acc1;
  #pragma unroll
  for (int i = 0; i < 8; ++i) { acc0.e[i] = 0.0f; acc1.e[i] = 0.0f; }

  for (int k0 = 0; k0 < K; k0 += 32) {
    // ---- stage A (64x32) with vector loads + 128-bit LDS store ----
    {
      const float* pa = A + (size_t)(bm + arow) * K + (k0 + acol);
      const float4 a0 = *(const float4*)(pa);
      const float4 a1 = *(const float4*)(pa + 4);
      U8H va;
      va.e[0] = (_Float16)a0.x; va.e[1] = (_Float16)a0.y;
      va.e[2] = (_Float16)a0.z; va.e[3] = (_Float16)a0.w;
      va.e[4] = (_Float16)a1.x; va.e[5] = (_Float16)a1.y;
      va.e[6] = (_Float16)a1.z; va.e[7] = (_Float16)a1.w;
      *(v8h*)&As[arow][acol] = va.v;
    }
    // ---- stage B (32x64) transposed into Bst[n][k] ----
    {
      const float* pb = B + (size_t)(k0 + brow) * N + (bn + bcol);
      const float4 b0 = *(const float4*)(pb);
      const float4 b1 = *(const float4*)(pb + 4);
      Bst[bcol + 0][brow] = (_Float16)b0.x;
      Bst[bcol + 1][brow] = (_Float16)b0.y;
      Bst[bcol + 2][brow] = (_Float16)b0.z;
      Bst[bcol + 3][brow] = (_Float16)b0.w;
      Bst[bcol + 4][brow] = (_Float16)b1.x;
      Bst[bcol + 5][brow] = (_Float16)b1.y;
      Bst[bcol + 6][brow] = (_Float16)b1.z;
      Bst[bcol + 7][brow] = (_Float16)b1.w;
    }
    if (k0 + 32 < K) {
      __builtin_prefetch(A + (size_t)(bm + arow) * K + (k0 + 32 + acol), 0, 1);
      __builtin_prefetch(B + (size_t)(k0 + 32 + brow) * N + (bn + bcol), 0, 1);
    }
    __syncthreads();

    // ---- fragments: all contiguous 128-bit LDS loads ----
    U16H a, b0, b1;
    a.h[0] = *(const v8h*)&As[wr + col][hf * 16];       // K 0-7 / 16-23
    a.h[1] = *(const v8h*)&As[wr + col][hf * 16 + 8];   // K 8-15 / 24-31
    b0.v = *(const v16h*)&Bst[wc + col][hf * 16];       // K 0-15 / 16-31
    b1.v = *(const v16h*)&Bst[wc + 16 + col][hf * 16];
    __syncthreads();

    acc0.v = __builtin_amdgcn_wmma_f32_16x16x32_f16(false, a.v, false, b0.v,
                                                    (short)0, acc0.v, false, false);
    acc1.v = __builtin_amdgcn_wmma_f32_16x16x32_f16(false, a.v, false, b1.v,
                                                    (short)0, acc1.v, false, false);
  }

  #pragma unroll
  for (int r = 0; r < 8; ++r) {
    const size_t row = (size_t)(bm + wr + r + hf * 8) * N;
    C[row + bn + wc + col]      = acc0.e[r];
    C[row + bn + wc + 16 + col] = acc1.e[r];
  }
}

// ---------------------------------------------------------------------------
// RMS-norm + m-RoPE (Q,K) and f16 conversion (V). One wave per (token, head).
// Q/K written head-major [b,head,t,h]; V written TRANSPOSED [b,kv,h,t] so the
// attention P*V B-fragments are contiguous 32B global loads.
// ---------------------------------------------------------------------------
__global__ __launch_bounds__(32) void normrope_kernel(
    const float* __restrict__ Qf, const float* __restrict__ Kf,
    const float* __restrict__ Vf, const float* __restrict__ qw,
    const float* __restrict__ kw, const int* __restrict__ pos,
    _Float16* __restrict__ Qh, _Float16* __restrict__ Kh,
    _Float16* __restrict__ Vt)
{
  const int tok  = blockIdx.x;          // b*T + t
  const int b    = tok / TQ;
  const int t    = tok % TQ;
  const int hd   = blockIdx.y;          // 0..31 Q, 32..39 K, 40..47 V
  const int lane = threadIdx.x;

  const float* src;
  const float* w;
  _Float16* dst;
  if (hd < NQ) {
    src = Qf + (size_t)tok * (NQ * HQ) + hd * HQ;
    dst = Qh + ((size_t)(b * NQ + hd) * TQ + t) * HQ;
    w   = qw;
  } else if (hd < NQ + KHQ) {
    const int kvh = hd - NQ;
    src = Kf + (size_t)tok * (KHQ * HQ) + kvh * HQ;
    dst = Kh + ((size_t)(b * KHQ + kvh) * TQ + t) * HQ;
    w   = kw;
  } else {
    const int kvh = hd - NQ - KHQ;
    src = Vf + (size_t)tok * (KHQ * HQ) + kvh * HQ;
    _Float16* vb = Vt + (size_t)(b * KHQ + kvh) * HQ * TQ;
    #pragma unroll
    for (int i = 0; i < 4; ++i) {
      const int h = lane + 32 * i;
      vb[(size_t)h * TQ + t] = (_Float16)src[h];     // transposed store
    }
    return;
  }

  float v[4];
  float ss = 0.0f;
  #pragma unroll
  for (int i = 0; i < 4; ++i) {
    v[i] = src[lane + 32 * i];
    ss += v[i] * v[i];
  }
  #pragma unroll
  for (int m = 1; m < 32; m <<= 1) ss += __shfl_xor(ss, m, 32);
  const float rinv = rsqrtf(ss * (1.0f / HQ) + 1e-6f);
  #pragma unroll
  for (int i = 0; i < 4; ++i)
    v[i] = w[lane + 32 * i] * v[i] * rinv;

  #pragma unroll
  for (int p = 0; p < 2; ++p) {
    const int i = lane + 32 * p;        // rotary pair 0..63
    int axis = 0;                       // mrope_section (24,20,20)
    if (i < 60) {
      const int m3 = i % 3;
      axis = (m3 == 1) ? 1 : (m3 == 2) ? 2 : 0;
    }
    const float position = (float)pos[axis * (BQ * TQ) + tok];
    const float inv_ts = __powf(1.0e6f, -(float)i * (1.0f / 64.0f));
    float s, c;
    __sincosf(position * inv_ts, &s, &c);
    const float f  = v[p];
    const float se = v[p + 2];
    dst[i]      = (_Float16)(f * c - se * s);
    dst[i + 64] = (_Float16)(se * c + f * s);
  }
}

// ---------------------------------------------------------------------------
// TDM: async-load one 32x128 f16 K tile (row-major) from global into LDS.
// D# per CDNA5 ISA: group0 = {count=1, lds_addr,
// global_addr, type=2}; group1 = {data_size=2B, tensor_dim0=128,
// tensor_dim1=32, tile 128x32, stride0=128}. Groups 2/3 zero (2D tensor).
// ---------------------------------------------------------------------------
#if USE_TDM
__device__ __forceinline__ void tdm_load_k_tile(const _Float16* g, unsigned lds_off)
{
  const unsigned long long ga = (unsigned long long)(uintptr_t)g;
  const u32x4 g0 = { 1u,                                     // count=1
                     lds_off,                                // lds_addr
                     (unsigned)(ga & 0xffffffffu),           // global_addr lo
                     (unsigned)((ga >> 32) & 0x1ffffffu) | 0x80000000u }; // hi | type=2
  const i32x8 g1 = { (int)0x10000u,          // data_size = 1 (2 bytes)
                     (int)(128u << 16),      // tensor_dim0 = 128
                     (int)(32u << 16),       // tensor_dim1 = 32
                     (int)(128u << 16),      // tile_dim0 = 128
                     32,                     // tile_dim1 = 32
                     128,                    // tensor_dim0_stride = 128
                     0, 0 };
  const i32x4 z4 = { 0, 0, 0, 0 };
#if __clang_major__ >= 23
  const i32x8 z8 = { 0, 0, 0, 0, 0, 0, 0, 0 };
  __builtin_amdgcn_tensor_load_to_lds(g0, g1, z4, z4, z8, 0);
#else
  __builtin_amdgcn_tensor_load_to_lds(g0, g1, z4, z4, 0);
#endif
}
#endif

// ---------------------------------------------------------------------------
// Flash attention: one wave per (16 query rows, head, batch). Keys in chunks
// of 32 with online softmax. K tiles TDM double-buffered through LDS
// (s_wait_tensorcnt), V read directly from its transposed global layout.
// ---------------------------------------------------------------------------
__global__ __launch_bounds__(32) void attn_kernel(
    const _Float16* __restrict__ Qh, const _Float16* __restrict__ Kh,
    const _Float16* __restrict__ Vt, float* __restrict__ Oqkv)
{
  // Kt declared first -> LDS offset 0 (TDM descriptor assumes this).
  __shared__ __align__(32) _Float16 Kt[2][32][HQ];  // 16 KB double buffer
  __shared__ __align__(32) _Float16 Pt[16][32];     // probabilities

  const int lane = threadIdx.x;
  const int col  = lane & 15;
  const int hf   = lane >> 4;
  const int q0   = blockIdx.x * 16;
  const int n    = blockIdx.y;
  const int b    = blockIdx.z;
  const int kvh  = n >> 2;                          // GQA group of 4

  const _Float16* Qb  = Qh + ((size_t)(b * NQ + n) * TQ) * HQ;
  const _Float16* Kb  = Kh + ((size_t)(b * KHQ + kvh) * TQ) * HQ;
  const _Float16* Vtb = Vt + (size_t)(b * KHQ + kvh) * HQ * TQ; // [h][t]

  // Q A-fragments, 16x128 as 4 chunks of K=32
  U16H qf[4];
  #pragma unroll
  for (int c = 0; c < 4; ++c) {
    const _Float16* base = Qb + (size_t)(q0 + col) * HQ + 32 * c + hf * 16;
    qf[c].h[0] = *(const v8h*)(base);
    qf[c].h[1] = *(const v8h*)(base + 8);
  }

  U8F oacc[8];
  float mrun[8], lrun[8];
  #pragma unroll
  for (int r = 0; r < 8; ++r) {
    mrun[r] = -3.0e38f;
    lrun[r] = 0.0f;
    #pragma unroll
    for (int i = 0; i < 8; ++i) oacc[r].e[i] = 0.0f;
  }

  const float scale = 0.08838834764831845f;         // 1/sqrt(128)
  const int nchunks = (q0 + 16 + 31) >> 5;          // causal bound, 32-key chunks

#if USE_TDM
  tdm_load_k_tile(Kb, 0u);                          // prime buffer 0
#endif

  for (int ci = 0; ci < nchunks; ++ci) {
    const int kk  = ci * 32;
#if USE_TDM
    const int buf = ci & 1;
    if (ci + 1 < nchunks) {                         // prefetch next K tile
      tdm_load_k_tile(Kb + (size_t)(kk + 32) * HQ,
                      (unsigned)(((ci + 1) & 1) * 32 * HQ * 2));
      __builtin_amdgcn_s_wait_tensorcnt((short)1);  // current buffer ready
    } else {
      __builtin_amdgcn_s_wait_tensorcnt((short)0);
    }
#endif

    // ---- scores S = Q * K^T (two 16x16 tiles) ----
    U8F s[2];
    #pragma unroll
    for (int sub = 0; sub < 2; ++sub) {
      #pragma unroll
      for (int i = 0; i < 8; ++i) s[sub].e[i] = 0.0f;
      #pragma unroll
      for (int c = 0; c < 4; ++c) {
#if USE_TDM
        const v16h kf = *(const v16h*)&Kt[buf][sub * 16 + col][32 * c + hf * 16];
#else
        const v16h kf = *(const v16h*)(Kb + (size_t)(kk + sub * 16 + col) * HQ +
                                       32 * c + hf * 16);
#endif
        s[sub].v = __builtin_amdgcn_wmma_f32_16x16x32_f16(false, qf[c].v, false, kf,
                                                          (short)0, s[sub].v, false, false);
      }
    }

    // ---- scale, causal mask, online softmax ----
    #pragma unroll
    for (int r = 0; r < 8; ++r) {
      const int qrow = q0 + r + hf * 8;
      #pragma unroll
      for (int sub = 0; sub < 2; ++sub) {
        const int key = kk + sub * 16 + col;
        const float sv = s[sub].e[r] * scale;
        s[sub].e[r] = (key <= qrow) ? sv : K_MASK_F;
      }
      float cm = fmaxf(s[0].e[r], s[1].e[r]);
      #pragma unroll
      for (int m = 1; m < 16; m <<= 1) cm = fmaxf(cm, __shfl_xor(cm, m, 32));
      const float mnew = fmaxf(mrun[r], cm);
      const float corr = __expf(mrun[r] - mnew);
      mrun[r] = mnew;
      float rsum = 0.0f;
      #pragma unroll
      for (int sub = 0; sub < 2; ++sub) {
        const float p = __expf(s[sub].e[r] - mnew);
        s[sub].e[r] = p;
        rsum += p;
      }
      #pragma unroll
      for (int m = 1; m < 16; m <<= 1) rsum += __shfl_xor(rsum, m, 32);
      lrun[r] = lrun[r] * corr + rsum;
      #pragma unroll
      for (int ht = 0; ht < 8; ++ht) oacc[ht].e[r] *= corr;
      Pt[r + hf * 8][col]      = (_Float16)s[0].e[r];
      Pt[r + hf * 8][16 + col] = (_Float16)s[1].e[r];
    }
    __syncthreads();

    // P as A-fragment (16 queries x 32 keys), contiguous 128-bit LDS loads
    U16H pf;
    pf.h[0] = *(const v8h*)&Pt[col][hf * 16];
    pf.h[1] = *(const v8h*)&Pt[col][hf * 16 + 8];

    // O += P * V: V B-fragments straight from transposed global layout
    #pragma unroll
    for (int ht = 0; ht < 8; ++ht) {
      const v16h vf = *(const v16h*)(Vtb + (size_t)(ht * 16 + col) * TQ + kk + hf * 16);
      oacc[ht].v = __builtin_amdgcn_wmma_f32_16x16x32_f16(false, pf.v, false, vf,
                                                          (short)0, oacc[ht].v, false, false);
    }
    __syncthreads();
  }

  // ---- normalize and scatter to qkv buffer [B*T][N*H] ----
  #pragma unroll
  for (int r = 0; r < 8; ++r) {
    const float linv = 1.0f / lrun[r];
    const size_t row = (size_t)(b * TQ + q0 + r + hf * 8) * (NQ * HQ) + (size_t)n * HQ;
    #pragma unroll
    for (int ht = 0; ht < 8; ++ht)
      Oqkv[row + ht * 16 + col] = oacc[ht].e[r] * linv;
  }
}

// ---------------------------------------------------------------------------
// Launch: 3 projection GEMMs -> norm+rope -> flash attention -> output GEMM.
// Workspace (72 MB):
//   [ 0,32MB) Qf32 (2048x4096)  -- reused as qkv after norm/rope consumes it
//   [32,40MB) Kf32   [40,48MB) Vf32
//   [48,64MB) Q f16 head-major
//   [64,68MB) K f16 head-major
//   [68,72MB) V f16 transposed [b,kv,h,t]
// ---------------------------------------------------------------------------
extern "C" void kernel_launch(void* const* d_in, const int* in_sizes, int n_in,
                              void* d_out, int out_size, void* d_ws, size_t ws_size,
                              hipStream_t stream)
{
  (void)in_sizes; (void)n_in; (void)out_size; (void)ws_size;

  const float* x         = (const float*)d_in[0];
  const int*   positions = (const int*)d_in[1];
  // d_in[2] = attn_mask (causal, reconstructed analytically)
  const float* wq        = (const float*)d_in[3];
  const float* wk        = (const float*)d_in[4];
  const float* wv        = (const float*)d_in[5];
  const float* wo        = (const float*)d_in[6];
  const float* qw        = (const float*)d_in[7];
  const float* kw        = (const float*)d_in[8];
  float*       out       = (float*)d_out;

  uint8_t* ws = (uint8_t*)d_ws;
  const unsigned MT = BQ * TQ;                       // 2048 rows
  float*     Qf  = (float*)(ws);
  float*     Kf  = (float*)(ws + 32ull * 1024 * 1024);
  float*     Vf  = (float*)(ws + 40ull * 1024 * 1024);
  _Float16*  Qhp = (_Float16*)(ws + 48ull * 1024 * 1024);
  _Float16*  Khp = (_Float16*)(ws + 64ull * 1024 * 1024);
  _Float16*  Vtp = (_Float16*)(ws + 68ull * 1024 * 1024);
  float*     qkv = Qf;                               // reuse Qf32 region

  gemm_wmma_f16<<<dim3((NQ * HQ) / 64, MT / 64), 256, 0, stream>>>(
      x, wq, Qf, MT, NQ * HQ, DQ);
  gemm_wmma_f16<<<dim3((KHQ * HQ) / 64, MT / 64), 256, 0, stream>>>(
      x, wk, Kf, MT, KHQ * HQ, DQ);
  gemm_wmma_f16<<<dim3((KHQ * HQ) / 64, MT / 64), 256, 0, stream>>>(
      x, wv, Vf, MT, KHQ * HQ, DQ);

  normrope_kernel<<<dim3(MT, NQ + 2 * KHQ), 32, 0, stream>>>(
      Qf, Kf, Vf, qw, kw, positions, Qhp, Khp, Vtp);

  attn_kernel<<<dim3(TQ / 16, NQ, BQ), 32, 0, stream>>>(Qhp, Khp, Vtp, qkv);

  gemm_wmma_f16<<<dim3(DQ / 64, MT / 64), 256, 0, stream>>>(
      qkv, wo, out, MT, DQ, NQ * HQ);
}
